// ChebDecoder_26061861552299
// MI455X (gfx1250) — matrix operand order repaired and checked by reading, not compiled, not run on hardware
//
#include <hip/hip_runtime.h>
#include <math.h>

// ---------------------------------------------------------------------------
// ChebDecoder forward for MI455X (gfx1250, wave32, WMMA).
// Activations live in f16 (halved HBM traffic); all math f32.
// GEMMs: v_wmma_f32_16x16x32_f16 with A/B fragments loaded DIRECTLY from
// global memory via 16-byte loads for A, 2-byte strided loads for B (no LDS staging, no barriers):
//   A 16x32 frag: lane=(half,m16) -> rows m16, k-groups [8*half,+8) and
//                 [16+8*half,+8) of the k-step  (two global_load_b128)
//   B 32x16 frag: lane=(half,n16) -> column n16, same k-groups as A (16 strided halves)
// K loop iterates per Chebyshev term with pure pointer increments (no
// bounds checks / term selects) whenever Cterm % 32 == 0 (all GEMMs except
// block5's K=48/K=16, which take a guarded fallback).
// Internal activation layout: (B, N, C) channel-fastest, f16.
// ---------------------------------------------------------------------------

typedef __attribute__((ext_vector_type(4)))  _Float16 v4h;
typedef __attribute__((ext_vector_type(8)))  _Float16 v8h;
typedef __attribute__((ext_vector_type(16))) _Float16 v16h;
typedef __attribute__((ext_vector_type(8)))  float    v8f;

#define EW_BLOCK 256

// ---------------------------------------------------------------------------
// f32 -> f16 weight conversion (once per call, ~1.1M elements total)
// ---------------------------------------------------------------------------
__global__ __launch_bounds__(EW_BLOCK)
void f32_to_f16_kernel(const float* __restrict__ src, _Float16* __restrict__ dst, int n)
{
    const int i = blockIdx.x * EW_BLOCK + threadIdx.x;
    if (i < n) dst[i] = (_Float16)src[i];
}

// ---------------------------------------------------------------------------
// unpool(prev) ++ enc -> X (B, Nn, Cin) f16, vectorized by 4 channels.
// ---------------------------------------------------------------------------
__global__ __launch_bounds__(EW_BLOCK)
void concat_unpool_kernel(const float* __restrict__ prevF32, const _Float16* __restrict__ prevF16,
                          int prevCM, const float* __restrict__ enc,
                          _Float16* __restrict__ X, int Nn, int Cprev, int Cenc)
{
    const int Cin = Cprev + Cenc;
    const int C4 = Cin >> 2;
    const size_t total = (size_t)2 * Nn * C4;
    const size_t idx = (size_t)blockIdx.x * EW_BLOCK + threadIdx.x;
    if (idx >= total) return;
    const int c4 = (int)(idx % C4) << 2;
    const size_t t = idx / C4;
    const int n = (int)(t % Nn);
    const int b = (int)(t / Nn);
    v4h o;
    if (c4 < Cprev) {
        const int Vn = Nn / 6;
        const int l = n / Vn;
        const int j = n % Vn;
        const int Nprev = Nn >> 1;
        const int np = l * (Vn >> 1) + (j >> 1);
        if (prevCM) {
#pragma unroll
            for (int i = 0; i < 4; ++i)
                o[i] = (_Float16)prevF32[((size_t)b * Cprev + c4 + i) * Nprev + np];
        } else {
            o = *reinterpret_cast<const v4h*>(prevF16 + ((size_t)b * Nprev + np) * Cprev + c4);
        }
    } else {
        const int ce = c4 - Cprev;
#pragma unroll
        for (int i = 0; i < 4; ++i)
            o[i] = (_Float16)enc[((size_t)b * Cenc + ce + i) * Nn + n];
    }
    *reinterpret_cast<v4h*>(X + idx * 4) = o;
}

// ---------------------------------------------------------------------------
// SpMM: Y = L @ X (fixed degree 8; node n owns edges [8n,8n+8)).
// X,Y: (B, Nn, C) f16; 4 channels per thread, f32 accumulation.
// ---------------------------------------------------------------------------
__global__ __launch_bounds__(EW_BLOCK)
void spmm_kernel(const _Float16* __restrict__ X,
                 const int* __restrict__ cols, const float* __restrict__ vals,
                 _Float16* __restrict__ Y, int Nn, int C)
{
    const int C4 = C >> 2;
    const size_t total = (size_t)2 * Nn * C4;
    const size_t idx = (size_t)blockIdx.x * EW_BLOCK + threadIdx.x;
    if (idx >= total) return;
    const int c4 = (int)(idx % C4) << 2;
    const size_t t = idx / C4;
    const int n = (int)(t % Nn);
    const int b = (int)(t / Nn);
    const _Float16* Xb = X + (size_t)b * Nn * C;
    const int e0 = n * 8;
    float a0 = 0.f, a1 = 0.f, a2 = 0.f, a3 = 0.f;
#pragma unroll
    for (int j = 0; j < 8; ++j) {
        const int col = cols[e0 + j];
        const float w = vals[e0 + j];
        const v4h x = *reinterpret_cast<const v4h*>(Xb + (size_t)col * C + c4);
        a0 = fmaf(w, (float)x[0], a0); a1 = fmaf(w, (float)x[1], a1);
        a2 = fmaf(w, (float)x[2], a2); a3 = fmaf(w, (float)x[3], a3);
    }
    v4h o; o[0] = (_Float16)a0; o[1] = (_Float16)a1; o[2] = (_Float16)a2; o[3] = (_Float16)a3;
    *reinterpret_cast<v4h*>(Y + idx * 4) = o;
}

// X2 = 2*(L @ X1) - X0
__global__ __launch_bounds__(EW_BLOCK)
void spmm_cheb_kernel(const _Float16* __restrict__ X1, const _Float16* __restrict__ X0,
                      const int* __restrict__ cols, const float* __restrict__ vals,
                      _Float16* __restrict__ X2, int Nn, int C)
{
    const int C4 = C >> 2;
    const size_t total = (size_t)2 * Nn * C4;
    const size_t idx = (size_t)blockIdx.x * EW_BLOCK + threadIdx.x;
    if (idx >= total) return;
    const int c4 = (int)(idx % C4) << 2;
    const size_t t = idx / C4;
    const int n = (int)(t % Nn);
    const int b = (int)(t / Nn);
    const _Float16* Xb = X1 + (size_t)b * Nn * C;
    const int e0 = n * 8;
    float a0 = 0.f, a1 = 0.f, a2 = 0.f, a3 = 0.f;
#pragma unroll
    for (int j = 0; j < 8; ++j) {
        const int col = cols[e0 + j];
        const float w = vals[e0 + j];
        const v4h x = *reinterpret_cast<const v4h*>(Xb + (size_t)col * C + c4);
        a0 = fmaf(w, (float)x[0], a0); a1 = fmaf(w, (float)x[1], a1);
        a2 = fmaf(w, (float)x[2], a2); a3 = fmaf(w, (float)x[3], a3);
    }
    const v4h x0 = *reinterpret_cast<const v4h*>(X0 + idx * 4);
    v4h o;
    o[0] = (_Float16)(2.f * a0 - (float)x0[0]);
    o[1] = (_Float16)(2.f * a1 - (float)x0[1]);
    o[2] = (_Float16)(2.f * a2 - (float)x0[2]);
    o[3] = (_Float16)(2.f * a3 - (float)x0[3]);
    *reinterpret_cast<v4h*>(X2 + idx * 4) = o;
}

// ---------------------------------------------------------------------------
// WMMA GEMM helpers
// ---------------------------------------------------------------------------
static __device__ __forceinline__ v16h cat16(v8h lo, v8h hi)
{
    return __builtin_shufflevector(lo, hi, 0, 1, 2, 3, 4, 5, 6, 7,
                                   8, 9, 10, 11, 12, 13, 14, 15);
}

static __device__ __forceinline__ v8h zero8()
{
    v8h z;
#pragma unroll
    for (int i = 0; i < 8; ++i) z[i] = (_Float16)0.f;
    return z;
}

// One Chebyshev term, Cterm % 32 == 0: pure pointer-increment hot loop.
// arow points at this lane's A row (k-offset 8*half already applied);
// wp is this lane's weight-row pointer and is advanced across the term.
template <int NT>
static __device__ __forceinline__ void mm_term(v8f* acc, const _Float16* __restrict__ arow,
                                               const _Float16*& wp, int Cterm, int Cout)
{
    for (int ks = 0; ks < Cterm; ks += 32) {
        const v16h a = cat16(*reinterpret_cast<const v8h*>(arow + ks),
                             *reinterpret_cast<const v8h*>(arow + ks + 16));
#pragma unroll
        for (int t = 0; t < NT; ++t) {
            v16h b;
#pragma unroll
            for (int s = 0; s < 16; ++s)
                b[s] = wp[(size_t)(s < 8 ? s : s + 8) * Cout + t * 16];
            acc[t] = __builtin_amdgcn_wmma_f32_16x16x32_f16(
                false, a, false, b, (short)0, acc[t], false, false);
        }
        wp += (size_t)32 * Cout;
    }
}

static __device__ __forceinline__ v8h load_a8_guard(const _Float16* __restrict__ T0,
                                                    const _Float16* __restrict__ T1,
                                                    const _Float16* __restrict__ T2,
                                                    int Cterm, int Kd, int m, int ks)
{
    if (ks < Kd) {
        const int sel = (ks >= Cterm) ? ((ks >= 2 * Cterm) ? 2 : 1) : 0;
        const _Float16* src = (sel == 0) ? T0 : ((sel == 1) ? T1 : T2);
        return *reinterpret_cast<const v8h*>(src + (size_t)m * Cterm + (ks - sel * Cterm));
    }
    return zero8();
}

// GEMM: C[m,o] = sum_k A[m,k]*W[k,o] + bias[o]; A = nterms buffers (M,Cterm)
// concatenated along K.  8 waves/block, wave tile 16 x (16*NT);
// grid = (M/128, Cout/(16*NT)).
template <int NT>
__global__ __launch_bounds__(256)
void gemm_wmma_kernel(const _Float16* __restrict__ T0, const _Float16* __restrict__ T1,
                      const _Float16* __restrict__ T2, int nterms, int Cterm,
                      const _Float16* __restrict__ Wf, const float* __restrict__ bias,
                      _Float16* __restrict__ Cmat, int Cout)
{
    const int lane = threadIdx.x & 31;
    const int wave = threadIdx.x >> 5;
    const int half = lane >> 4;
    const int mBase = blockIdx.x * 128 + wave * 16;
    const int mA = mBase + (lane & 15);          // A-fragment row for this lane
    const int oBase = blockIdx.y * (16 * NT);

    v8f acc[NT];
#pragma unroll
    for (int t = 0; t < NT; ++t)
#pragma unroll
        for (int r = 0; r < 8; ++r) acc[t][r] = 0.f;

    if ((Cterm & 31) == 0) {
        // hot path: no bounds checks, no term selects
        const _Float16* wp = Wf + (size_t)(8 * half) * Cout + oBase + (lane & 15);
        const size_t aoff = (size_t)mA * Cterm + 8 * half;
        mm_term<NT>(acc, T0 + aoff, wp, Cterm, Cout);
        if (nterms == 3) {
            mm_term<NT>(acc, T1 + aoff, wp, Cterm, Cout);
            mm_term<NT>(acc, T2 + aoff, wp, Cterm, Cout);
        }
    } else {
        // fallback (Cterm == 16): guarded, group-granular zero fill
        const int Kd = nterms * Cterm;
        for (int kk = 0; kk < Kd; kk += 32) {
            const v16h a = cat16(
                load_a8_guard(T0, T1, T2, Cterm, Kd, mA, kk + 8 * half),
                load_a8_guard(T0, T1, T2, Cterm, Kd, mA, kk + 16 + 8 * half));
            const int kb = kk + 8 * half;
            const _Float16* wcol = Wf + oBase + (lane & 15);
#pragma unroll
            for (int t = 0; t < NT; ++t) {
                v16h b;
#pragma unroll
                for (int s = 0; s < 16; ++s) {
                    const int kr = kb + (s < 8 ? s : s + 8);
                    b[s] = (kr < Kd) ? wcol[(size_t)kr * Cout + t * 16] : (_Float16)0.f;
                }
                acc[t] = __builtin_amdgcn_wmma_f32_16x16x32_f16(
                    false, a, false, b, (short)0, acc[t], false, false);
            }
        }
    }

    // D layout: lane = 16*(m>=8) + n16, VGPR r = low 3 bits of m
    const int n16 = lane & 15;
    const int mh = half * 8;
#pragma unroll
    for (int t = 0; t < NT; ++t) {
        const int o = oBase + t * 16 + n16;
        const float bv = bias[o];
#pragma unroll
        for (int r = 0; r < 8; ++r)
            Cmat[(size_t)(mBase + mh + r) * Cout + o] = (_Float16)(acc[t][r] + bv);
    }
}

// ---------------------------------------------------------------------------
// BatchNorm statistics: deterministic two-pass (no float atomics).
// X: (M, C) f16, C in {16,32,64,128,256}, 256 % C == 0.
// ---------------------------------------------------------------------------
__global__ __launch_bounds__(256)
void bn_stats_partial_kernel(const _Float16* __restrict__ X, float* __restrict__ part,
                             int M, int C, int iter)
{
    const int tid = threadIdx.x;
    const int c = tid % C;
    const int rl = tid / C;
    const int tpc = 256 / C;
    const int r0 = blockIdx.x * (iter * tpc);
    float s = 0.f, ss = 0.f;
    for (int i = 0; i < iter; ++i) {
        const int r = r0 + rl + i * tpc;
        if (r < M) {
            const float v = (float)X[(size_t)r * C + c];
            s += v; ss += v * v;
        }
    }
    __shared__ float sh[512];
    sh[tid] = s; sh[256 + tid] = ss;
    __syncthreads();
    if (rl == 0) {
        float S = 0.f, SS = 0.f;
        for (int k = 0; k < tpc; ++k) { S += sh[c + k * C]; SS += sh[256 + c + k * C]; }
        part[(size_t)blockIdx.x * 2 * C + c]     = S;
        part[(size_t)blockIdx.x * 2 * C + C + c] = SS;
    }
}

__global__ __launch_bounds__(256)
void bn_reduce_kernel(const float* __restrict__ part, float* __restrict__ red,
                      int nb, int C)
{
    const int c = blockIdx.x * 256 + threadIdx.x;
    if (c >= C) return;
    float S = 0.f, SS = 0.f;
    for (int b = 0; b < nb; ++b) {
        S  += part[(size_t)b * 2 * C + c];
        SS += part[(size_t)b * 2 * C + C + c];
    }
    red[c] = S; red[C + c] = SS;
}

// h = relu(bn(x)) in place, 4 channels/thread
__global__ __launch_bounds__(EW_BLOCK)
void bn_apply_relu_kernel(_Float16* __restrict__ X, const float* __restrict__ red,
                          const float* __restrict__ g, const float* __restrict__ be,
                          int M, int C)
{
    const int C4 = C >> 2;
    const size_t total = (size_t)M * C4;
    const size_t idx = (size_t)blockIdx.x * EW_BLOCK + threadIdx.x;
    if (idx >= total) return;
    const int c4 = (int)(idx % C4) << 2;
    const float invM = 1.f / (float)M;
    v4h x = *reinterpret_cast<const v4h*>(X + idx * 4);
    v4h o;
#pragma unroll
    for (int i = 0; i < 4; ++i) {
        const int c = c4 + i;
        const float mean = red[c] * invM;
        const float var = red[C + c] * invM - mean * mean;
        const float rstd = rsqrtf(var + 1e-5f);
        const float v = ((float)x[i] - mean) * rstd * g[c] + be[c];
        o[i] = (_Float16)(v > 0.f ? v : 0.f);
    }
    *reinterpret_cast<v4h*>(X + idx * 4) = o;
}

// d = relu(bn(h2raw) + res), 4 channels/thread
__global__ __launch_bounds__(EW_BLOCK)
void bn_add_relu_kernel(const _Float16* __restrict__ H, const _Float16* __restrict__ RES,
                        const float* __restrict__ red, const float* __restrict__ g,
                        const float* __restrict__ be, _Float16* __restrict__ D,
                        int M, int C)
{
    const int C4 = C >> 2;
    const size_t total = (size_t)M * C4;
    const size_t idx = (size_t)blockIdx.x * EW_BLOCK + threadIdx.x;
    if (idx >= total) return;
    const int c4 = (int)(idx % C4) << 2;
    const float invM = 1.f / (float)M;
    const v4h h = *reinterpret_cast<const v4h*>(H + idx * 4);
    const v4h rr = *reinterpret_cast<const v4h*>(RES + idx * 4);
    v4h o;
#pragma unroll
    for (int i = 0; i < 4; ++i) {
        const int c = c4 + i;
        const float mean = red[c] * invM;
        const float var = red[C + c] * invM - mean * mean;
        const float rstd = rsqrtf(var + 1e-5f);
        const float v = ((float)h[i] - mean) * rstd * g[c] + be[c] + (float)rr[i];
        o[i] = (_Float16)(v > 0.f ? v : 0.f);
    }
    *reinterpret_cast<v4h*>(D + idx * 4) = o;
}

// ---------------------------------------------------------------------------
// Final: orientation max-pool (L=6) -> 16x10 projection -> log_softmax.
// D5: (B, 6*V5, 16) f16.  out: (B, 10, V5) f32 (reference layout).
// ---------------------------------------------------------------------------
__global__ __launch_bounds__(EW_BLOCK)
void final_kernel(const _Float16* __restrict__ D5, const float* __restrict__ Wc,
                  float* __restrict__ out, int V5)
{
    const int idx = blockIdx.x * EW_BLOCK + threadIdx.x;
    if (idx >= 2 * V5) return;
    const int v = idx % V5;
    const int b = idx / V5;
    const int N5 = 6 * V5;
    float mx[16];
#pragma unroll
    for (int c = 0; c < 16; ++c) mx[c] = -3.402823466e38f;
    for (int l = 0; l < 6; ++l) {
        const _Float16* p = D5 + ((size_t)b * N5 + (size_t)l * V5 + v) * 16;
        const v8h x0 = *reinterpret_cast<const v8h*>(p);
        const v8h x1 = *reinterpret_cast<const v8h*>(p + 8);
#pragma unroll
        for (int c = 0; c < 8; ++c) {
            const float a = (float)x0[c]; if (a > mx[c]) mx[c] = a;
            const float d = (float)x1[c]; if (d > mx[8 + c]) mx[8 + c] = d;
        }
    }
    float logit[10];
#pragma unroll
    for (int o = 0; o < 10; ++o) {
        float a = 0.f;
#pragma unroll
        for (int c = 0; c < 16; ++c) a = fmaf(mx[c], Wc[c * 10 + o], a);
        logit[o] = a;
    }
    float m = logit[0];
#pragma unroll
    for (int o = 1; o < 10; ++o) m = logit[o] > m ? logit[o] : m;
    float se = 0.f;
#pragma unroll
    for (int o = 0; o < 10; ++o) se += __expf(logit[o] - m);
    const float lse = m + __logf(se);
#pragma unroll
    for (int o = 0; o < 10; ++o)
        out[((size_t)b * 10 + o) * V5 + v] = logit[o] - lse;
}

// ---------------------------------------------------------------------------
// Host orchestration
// ---------------------------------------------------------------------------
struct BP { const float *W1, *W2, *Wres, *b1, *b2, *be1, *be2, *bres, *g1, *g2;
            const _Float16 *W1h, *W2h, *Wresh; };
struct GR { const int* cols; const float* vals; };

static inline int divUpHost(long long a, int b) { return (int)((a + b - 1) / b); }

static void launch_gemm(const _Float16* t0, const _Float16* t1, const _Float16* t2,
                        int nterms, int Cterm, const _Float16* Wf, const float* bias,
                        _Float16* Cmat, int M, int Cout, hipStream_t s)
{
    const int NT = (Cout >= 128) ? 8 : (Cout >= 64 ? 4 : (Cout >= 32 ? 2 : 1));
    dim3 grid(M / 128, Cout / (16 * NT));
    if (NT == 8)
        gemm_wmma_kernel<8><<<grid, 256, 0, s>>>(t0, t1, t2, nterms, Cterm, Wf, bias, Cmat, Cout);
    else if (NT == 4)
        gemm_wmma_kernel<4><<<grid, 256, 0, s>>>(t0, t1, t2, nterms, Cterm, Wf, bias, Cmat, Cout);
    else if (NT == 2)
        gemm_wmma_kernel<2><<<grid, 256, 0, s>>>(t0, t1, t2, nterms, Cterm, Wf, bias, Cmat, Cout);
    else
        gemm_wmma_kernel<1><<<grid, 256, 0, s>>>(t0, t1, t2, nterms, Cterm, Wf, bias, Cmat, Cout);
}

static void launch_bn_stats(const _Float16* X, float* part, float* red,
                            int M, int C, hipStream_t s)
{
    const int iter = 16;
    const int rowsPerBlk = iter * (256 / C);
    const int nb = divUpHost(M, rowsPerBlk);
    bn_stats_partial_kernel<<<nb, 256, 0, s>>>(X, part, M, C, iter);
    bn_reduce_kernel<<<divUpHost(C, 256), 256, 0, s>>>(part, red, nb, C);
}

static void run_block(const float* prevF32, const _Float16* prevF16, int Cprev,
                      const float* enc, int Cenc,
                      int Nn, int Cout, const BP& p, const GR& g,
                      _Float16* X, _Float16* X1a, _Float16* X2a, _Float16* H1,
                      _Float16* Dst, float* part, float* red, hipStream_t s)
{
    const int Cin = Cprev + Cenc;
    const int M = 2 * Nn;
    const size_t HOFF = 3145728;            // Cout-buffer stride (halves) = 6 MB
    _Float16* Hs1 = X1a;
    _Float16* Hs2 = X1a + HOFF;
    _Float16* H2  = X1a + 2 * HOFF;
    _Float16* RES = X2a;

    const long long q4in = (long long)M * (Cin >> 2);
    const long long q4out = (long long)M * (Cout >> 2);

    // 1. X = concat(unpool(prev), enc)
    concat_unpool_kernel<<<divUpHost(q4in, EW_BLOCK), EW_BLOCK, 0, s>>>(
        prevF32, prevF16, prevF32 != nullptr, enc, X, Nn, Cprev, Cenc);
    // 2-3. Chebyshev terms on X
    spmm_kernel<<<divUpHost(q4in, EW_BLOCK), EW_BLOCK, 0, s>>>(X, g.cols, g.vals, X1a, Nn, Cin);
    spmm_cheb_kernel<<<divUpHost(q4in, EW_BLOCK), EW_BLOCK, 0, s>>>(X1a, X, g.cols, g.vals, X2a, Nn, Cin);
    // 4. conv1 -> H1 (raw)
    launch_gemm(X, X1a, X2a, 3, Cin, p.W1h, p.b1, H1, M, Cout, s);
    // 5-6. BN + ReLU in place
    launch_bn_stats(H1, part, red, M, Cout, s);
    bn_apply_relu_kernel<<<divUpHost(q4out, EW_BLOCK), EW_BLOCK, 0, s>>>(H1, red, p.g1, p.be1, M, Cout);
    // 7-8. Chebyshev terms on H1 (X1/X2 arenas now free)
    spmm_kernel<<<divUpHost(q4out, EW_BLOCK), EW_BLOCK, 0, s>>>(H1, g.cols, g.vals, Hs1, Nn, Cout);
    spmm_cheb_kernel<<<divUpHost(q4out, EW_BLOCK), EW_BLOCK, 0, s>>>(Hs1, H1, g.cols, g.vals, Hs2, Nn, Cout);
    // 9. conv2 -> H2 (raw)
    launch_gemm(H1, Hs1, Hs2, 3, Cout, p.W2h, p.b2, H2, M, Cout, s);
    // 10. BN stats for conv2
    launch_bn_stats(H2, part, red, M, Cout, s);
    // 11. residual 1x1 GEMM (reads X, writes X2 arena)
    launch_gemm(X, X, X, 1, Cin, p.Wresh, p.bres, RES, M, Cout, s);
    // 12. d = relu(bn(H2) + RES)
    bn_add_relu_kernel<<<divUpHost(q4out, EW_BLOCK), EW_BLOCK, 0, s>>>(H2, RES, red, p.g2, p.be2, Dst, M, Cout);
}

extern "C" void kernel_launch(void* const* d_in, const int* in_sizes, int n_in,
                              void* d_out, int out_size, void* d_ws, size_t ws_size,
                              hipStream_t stream)
{
    (void)in_sizes; (void)n_in; (void)out_size; (void)ws_size;

    // ---- inputs -----------------------------------------------------------
    const float* xenc[6];
    for (int i = 0; i < 6; ++i) xenc[i] = (const float*)d_in[i];

    BP bp[6];
    for (int b = 1; b <= 5; ++b) {
        const int k = 6 + (b - 1) * 10;    // W1,W2,Wres,b1,b2,be1,be2,bres,g1,g2
        bp[b].W1 = (const float*)d_in[k + 0]; bp[b].W2 = (const float*)d_in[k + 1];
        bp[b].Wres = (const float*)d_in[k + 2];
        bp[b].b1 = (const float*)d_in[k + 3]; bp[b].b2 = (const float*)d_in[k + 4];
        bp[b].be1 = (const float*)d_in[k + 5]; bp[b].be2 = (const float*)d_in[k + 6];
        bp[b].bres = (const float*)d_in[k + 7];
        bp[b].g1 = (const float*)d_in[k + 8]; bp[b].g2 = (const float*)d_in[k + 9];
    }
    const float* Wconv = (const float*)d_in[56];

    GR gr[6];
    for (int i = 1; i <= 5; ++i) {
        const int k = 57 + (i - 1) * 3;    // cols, rows, vals
        gr[i].cols = (const int*)d_in[k + 0];
        gr[i].vals = (const float*)d_in[k + 2];
    }

    // ---- workspace arenas (bytes); activations are f16 --------------------
    char* ws = (char*)d_ws;
    _Float16* X   = (_Float16*)(ws + 0);          // 25,165,824 B (max M*Cin*2)
    _Float16* X1a = (_Float16*)(ws + 25165824);   // 25,165,824 B (also Hs1/Hs2/H2)
    _Float16* X2a = (_Float16*)(ws + 50331648);   // 25,165,824 B (also RES)
    _Float16* H1  = (_Float16*)(ws + 75497472);   //  6,291,456 B
    _Float16* D0  = (_Float16*)(ws + 81788928);   //  6,291,456 B
    _Float16* D1  = (_Float16*)(ws + 88080384);   //  6,291,456 B
    float* part   = (float*)(ws + 94371840);      //  2,097,152 B BN partials
    float* red    = (float*)(ws + 96468992);      //      2,048 B BN reduced stats
    _Float16* wf  = (_Float16*)(ws + 96471040);   //  ~2.27 MB f16 weights

    // ---- one-time (per call) f32 -> f16 weight conversion -----------------
    {
        const int cin[6]  = {0, 512, 512, 256, 128, 64};
        const int cout[6] = {0, 256, 128, 64, 32, 16};
        size_t off = 0;
        for (int b = 1; b <= 5; ++b) {
            const int n1 = 3 * cin[b] * cout[b];
            const int n2 = 3 * cout[b] * cout[b];
            const int n3 = cin[b] * cout[b];
            f32_to_f16_kernel<<<divUpHost(n1, EW_BLOCK), EW_BLOCK, 0, stream>>>(bp[b].W1, wf + off, n1);
            bp[b].W1h = wf + off; off += (size_t)n1;
            f32_to_f16_kernel<<<divUpHost(n2, EW_BLOCK), EW_BLOCK, 0, stream>>>(bp[b].W2, wf + off, n2);
            bp[b].W2h = wf + off; off += (size_t)n2;
            f32_to_f16_kernel<<<divUpHost(n3, EW_BLOCK), EW_BLOCK, 0, stream>>>(bp[b].Wres, wf + off, n3);
            bp[b].Wresh = wf + off; off += (size_t)n3;
        }
    }

    // ---- decoder blocks ---------------------------------------------------
    run_block(xenc[0], nullptr, 256, xenc[1], 256, 6144, 256, bp[1], gr[1],
              X, X1a, X2a, H1, D0, part, red, stream);
    run_block(nullptr, D0, 256, xenc[2], 256, 12288, 128, bp[2], gr[2],
              X, X1a, X2a, H1, D1, part, red, stream);
    run_block(nullptr, D1, 128, xenc[3], 128, 24576, 64, bp[3], gr[3],
              X, X1a, X2a, H1, D0, part, red, stream);
    run_block(nullptr, D0, 64, xenc[4], 64, 49152, 32, bp[4], gr[4],
              X, X1a, X2a, H1, D1, part, red, stream);
    run_block(nullptr, D1, 32, xenc[5], 32, 98304, 16, bp[5], gr[5],
              X, X1a, X2a, H1, D0, part, red, stream);

    // ---- pool over orientations, 16->10 conv, log_softmax ----------------
    const int V5 = 98304 / 6;   // 16384
    final_kernel<<<divUpHost(2 * V5, EW_BLOCK), EW_BLOCK, 0, stream>>>(
        D0, Wconv, (float*)d_out, V5);
}
